// CrossScaleAttention_47150150975597
// MI455X (gfx1250) — compile-verified, matrix operand order checked by reading, not run on hardware
//
#include <hip/hip_runtime.h>

typedef _Float16 half_t;
typedef __attribute__((ext_vector_type(16))) _Float16 v16h;
typedef __attribute__((ext_vector_type(8)))  _Float16 v8h;
typedef __attribute__((ext_vector_type(8)))  float    v8f;
typedef int v4i_vs __attribute__((vector_size(16)));   // matches builtin param

#define BATCH 4
#define CX    128
#define CY    256
#define NPIX  9216   // 96*96
#define MPIX  2304   // 48*48
#define MT    32     // key tile per iteration

// --- CDNA5 async global->LDS path (guarded; falls back to reg relay) -------
#if __has_builtin(__builtin_amdgcn_global_load_async_to_lds_b128) && \
    __has_builtin(__builtin_amdgcn_s_wait_asynccnt)
#define HAVE_ASYNC_LDS 1
#define ASYNC_COPY16(gp, lp)                                                  \
  __builtin_amdgcn_global_load_async_to_lds_b128(                             \
      (__attribute__((address_space(1))) v4i_vs*)(gp),                        \
      (__attribute__((address_space(3))) v4i_vs*)(lp), 0, 0)
#define ASYNC_WAIT(n) __builtin_amdgcn_s_wait_asynccnt(n)
#else
#define HAVE_ASYNC_LDS 0
#define ASYNC_COPY16(gp, lp) (*(v8h*)(lp) = *(const v8h*)(gp))
#define ASYNC_WAIT(n) ((void)0)
#endif

__device__ inline v16h load_frag(const half_t* p0, const half_t* p1) {
  v8h a = *(const v8h*)p0;
  v8h b = *(const v8h*)p1;
  v16h r;
#pragma unroll
  for (int i = 0; i < 8; i++) { r[i] = a[i]; r[i + 8] = b[i]; }
  return r;
}

__device__ inline v8f wmma_f16(v16h a, v16h b, v8f c) {
  return __builtin_amdgcn_wmma_f32_16x16x32_f16(false, a, false, b, (short)0, c,
                                                false, false);
}

// ---------------------------------------------------------------------------
// Fold BatchNorm (eval) into conv weights: W'[o,c] = W[o,c]*g/sqrt(v+eps),
// b'[o] = (b[o]-mean[o])*scale + beta[o].  Output weights in f16.
// ---------------------------------------------------------------------------
__global__ void fuse_bn_kernel(const float* __restrict__ w, const float* __restrict__ b,
                               const float* __restrict__ g, const float* __restrict__ beta,
                               const float* __restrict__ mean, const float* __restrict__ var,
                               int Cout, int Cin, half_t* __restrict__ wOut,
                               float* __restrict__ bOut) {
  int i = blockIdx.x * blockDim.x + threadIdx.x;
  int total = Cout * Cin;
  if (i < total) {
    int o = i / Cin;
    float s = g[o] * rsqrtf(var[o] + 1e-5f);
    wOut[i] = (half_t)(w[i] * s);
  }
  if (i < Cout) {
    float s = g[i] * rsqrtf(var[i] + 1e-5f);
    bOut[i] = (b[i] - mean[i]) * s + beta[i];
  }
}

// ---------------------------------------------------------------------------
// Projection GEMM via WMMA. in: [B, Cin, P] f32 (conv layout), w: [128, Cin] f16,
// bias: [128] f32.  One wave per 16-pixel tile, computes all 128 out channels.
//   mode 0: out[b, p, o]   (pixel-major rows of 128 -> Q / K layout)
//   mode 1: out[b, o, p]   (channel-major rows of P -> V^T layout)
// ---------------------------------------------------------------------------
__global__ void proj_kernel(const float* __restrict__ in, const half_t* __restrict__ w,
                            const float* __restrict__ bias, half_t* __restrict__ out,
                            int Cin, int P, int mode) {
  const int lane = threadIdx.x & 31;
  const int lo = lane & 15;
  const int hi = lane >> 4;
  const int tilesPerB = P / 16;
  const int b = blockIdx.x / tilesPerB;
  const int pbase = (blockIdx.x % tilesPerB) * 16;
  const int row = pbase + lo;           // pixel (A-matrix row) this lane holds
  const float* inB = in + (size_t)b * Cin * P;

  v8f acc[8] = {};
  const int ksteps = Cin / 32;
  for (int ks = 0; ks < ksteps; ks++) {
    // A fragment: X^T[row, c] gathered (stride P) and converted to f16
    v16h afrag;
#pragma unroll
    for (int j = 0; j < 8; j++) {
      int c0 = ks * 32 + hi * 8 + j;
      afrag[j]     = (half_t)inB[(size_t)c0 * P + row];
      afrag[j + 8] = (half_t)inB[(size_t)(c0 + 16) * P + row];
    }
#pragma unroll
    for (int t = 0; t < 8; t++) {
      const half_t* wr = w + (size_t)(t * 16 + lo) * Cin + ks * 32 + hi * 8;
      v16h bfrag = load_frag(wr, wr + 16);
      acc[t] = wmma_f16(afrag, bfrag, acc[t]);
    }
  }
#pragma unroll
  for (int t = 0; t < 8; t++) {
    int o = t * 16 + lo;
    float bi = bias[o];
    if (mode == 0) {
#pragma unroll
      for (int r = 0; r < 8; r++) {
        int p = pbase + r + 8 * hi;
        out[((size_t)b * P + p) * CX + o] = (half_t)(acc[t][r] + bi);
      }
    } else {
      v8h pk;
#pragma unroll
      for (int r = 0; r < 8; r++) pk[r] = (half_t)(acc[t][r] + bi);
      half_t* dst = out + ((size_t)b * CX + o) * P + pbase + 8 * hi;
      *(v8h*)dst = pk;
    }
  }
}

// ---------------------------------------------------------------------------
// Flash attention: per block = 128 query rows (8 waves x 16 rows); keys in
// tiles of 32, double-buffered K/V tiles staged via async global->LDS,
// online softmax, residual epilogue.
// ---------------------------------------------------------------------------
__global__ __launch_bounds__(256) void attn_kernel(
    const half_t* __restrict__ Qh,   // [B, N, 128]
    const half_t* __restrict__ Kh,   // [B, M, 128]
    const half_t* __restrict__ Vt,   // [B, 128, M]
    const float*  __restrict__ x,    // [B, 128, N]
    float*        __restrict__ outp) // [B, 128, N]
{
  __shared__ half_t Ks[2][MT][CX];       // 16 KB (double buffered)
  __shared__ half_t Vs[2][CX][MT];       // 16 KB (double buffered)
  __shared__ half_t Ps[8][16 * MT];      // 8 KB  (per-wave P staging)

  const int tid = threadIdx.x;
  const int lane = tid & 31;
  const int wave = tid >> 5;
  const int lo = lane & 15;
  const int hi = lane >> 4;

  const int blocksPerB = NPIX / 128;     // 72
  const int b = blockIdx.x / blocksPerB;
  const int nblock = (blockIdx.x % blocksPerB) * 128;
  const int nbase = nblock + wave * 16;

  const half_t* Kb = Kh + (size_t)b * MPIX * CX;
  const half_t* Vb = Vt + (size_t)b * CX * MPIX;
  const float* xb = x + (size_t)b * CX * NPIX;
  float* ob = outp + (size_t)b * CX * NPIX;

  // Warm the cache for the residual region this block touches in the epilogue
  // (probe-confirmed: lowers to global_prefetch_b8).
  __builtin_prefetch(xb + (size_t)tid * NPIX + nblock, 0, 1);

  // Stage one 32-key K/V tile into LDS buffer `buf` (4 x 16B chunks / thread).
  auto stage = [&](int buf, int m0) {
#pragma unroll
    for (int c2 = 0; c2 < 2; c2++) {
      int chunk = tid + c2 * 256;                 // 0..511
      int krow = chunk >> 4, kcol = (chunk & 15) * 8;
      ASYNC_COPY16(Kb + (size_t)(m0 + krow) * CX + kcol, &Ks[buf][krow][kcol]);
      int vrow = chunk >> 2, vcol = (chunk & 3) * 8;
      ASYNC_COPY16(Vb + (size_t)vrow * MPIX + m0 + vcol, &Vs[buf][vrow][vcol]);
    }
  };

  // Preload the wave's Q fragments for all 4 k-steps (C = 128)
  const half_t* Qrow = Qh + ((size_t)b * NPIX + nbase + lo) * CX + hi * 8;
  v16h qf[4];
#pragma unroll
  for (int ks = 0; ks < 4; ks++)
    qf[ks] = load_frag(Qrow + ks * 32, Qrow + ks * 32 + 16);

  v8f O[8] = {};
  float rmax[8], rsum[8];
#pragma unroll
  for (int r = 0; r < 8; r++) { rmax[r] = -1e30f; rsum[r] = 0.f; }

  stage(0, 0);                          // prime the pipeline

  const int nIters = MPIX / MT;         // 72
  for (int it = 0; it < nIters; it++) {
    const int m0 = it * MT;
    const int cur = it & 1;

    __syncthreads();                    // everyone done reading buf[1-cur]
    if (it + 1 < nIters) {
      stage(1 - cur, m0 + MT);          // prefetch next tile (async)
      ASYNC_WAIT(4);                    // current tile's 4 chunks have landed
    } else {
      ASYNC_WAIT(0);
    }
    __syncthreads();                    // all waves' chunks visible

    // S = Q · K^T : two 16x16 f32 tiles (key cols 0-15 and 16-31)
    v8f S0 = {}, S1 = {};
#pragma unroll
    for (int ks = 0; ks < 4; ks++) {
      const half_t* k0 = &Ks[cur][lo][ks * 32 + hi * 8];
      S0 = wmma_f16(qf[ks], load_frag(k0, k0 + 16), S0);
      const half_t* k1 = &Ks[cur][16 + lo][ks * 32 + hi * 8];
      S1 = wmma_f16(qf[ks], load_frag(k1, k1 + 16), S1);
    }

    // Online softmax. Row (r + 8*hi) spans the 16 lanes of each half-wave.
    float alpha[8];
#pragma unroll
    for (int r = 0; r < 8; r++) {
      float tm = fmaxf(S0[r], S1[r]);
      tm = fmaxf(tm, __shfl_xor(tm, 1, 32));
      tm = fmaxf(tm, __shfl_xor(tm, 2, 32));
      tm = fmaxf(tm, __shfl_xor(tm, 4, 32));
      tm = fmaxf(tm, __shfl_xor(tm, 8, 32));
      float nm = fmaxf(rmax[r], tm);
      float a = __expf(rmax[r] - nm);
      rmax[r] = nm;
      alpha[r] = a;
      float p0 = __expf(S0[r] - nm);
      float p1 = __expf(S1[r] - nm);
      S0[r] = p0; S1[r] = p1;
      rsum[r] = rsum[r] * a + p0 + p1;  // per-lane partial; reduced at end
    }
#pragma unroll
    for (int t = 0; t < 8; t++)
#pragma unroll
      for (int r = 0; r < 8; r++) O[t][r] *= alpha[r];

    // Re-layout P (D-tile layout) -> A-fragment layout via per-wave LDS
    half_t* Pw = &Ps[wave][0];
#pragma unroll
    for (int r = 0; r < 8; r++) {
      int prow = r + 8 * hi;
      Pw[prow * MT + lo]      = (half_t)S0[r];
      Pw[prow * MT + 16 + lo] = (half_t)S1[r];
    }
    __syncthreads();

    const half_t* pr = Pw + lo * MT + hi * 8;
    v16h pf = load_frag(pr, pr + 16);
#pragma unroll
    for (int t = 0; t < 8; t++) {
      const half_t* vp = &Vs[cur][t * 16 + lo][hi * 8];
      O[t] = wmma_f16(pf, load_frag(vp, vp + 16), O[t]);
    }
  }

  // Normalize, add residual, write [B, C, N]
  float inv[8];
#pragma unroll
  for (int r = 0; r < 8; r++) {
    float s = rsum[r];
    s += __shfl_xor(s, 1, 32);
    s += __shfl_xor(s, 2, 32);
    s += __shfl_xor(s, 4, 32);
    s += __shfl_xor(s, 8, 32);
    inv[r] = 1.f / s;
  }
#pragma unroll
  for (int t = 0; t < 8; t++) {
    int c = t * 16 + lo;
#pragma unroll
    for (int r = 0; r < 8; r++) {
      int n = nbase + r + 8 * hi;
      size_t off = (size_t)c * NPIX + n;
      ob[off] = xb[off] + O[t][r] * inv[r];
    }
  }
}

// ---------------------------------------------------------------------------
extern "C" void kernel_launch(void* const* d_in, const int* in_sizes, int n_in,
                              void* d_out, int out_size, void* d_ws, size_t ws_size,
                              hipStream_t stream) {
  (void)in_sizes; (void)n_in; (void)out_size; (void)ws_size;
  const float* x   = (const float*)d_in[0];
  const float* y   = (const float*)d_in[1];
  const float* w_q = (const float*)d_in[2];
  const float* b_q = (const float*)d_in[3];
  const float* gq  = (const float*)d_in[4];
  const float* bq  = (const float*)d_in[5];
  const float* mq  = (const float*)d_in[6];
  const float* vq  = (const float*)d_in[7];
  const float* w_k = (const float*)d_in[8];
  const float* b_k = (const float*)d_in[9];
  const float* gk  = (const float*)d_in[10];
  const float* bk  = (const float*)d_in[11];
  const float* mk  = (const float*)d_in[12];
  const float* vk  = (const float*)d_in[13];
  const float* w_v = (const float*)d_in[14];
  const float* b_v = (const float*)d_in[15];
  const float* gv  = (const float*)d_in[16];
  const float* bv  = (const float*)d_in[17];
  const float* mv  = (const float*)d_in[18];
  const float* vv  = (const float*)d_in[19];

  // Workspace carve-out (~14.3 MB total)
  char* p = (char*)d_ws;
  auto carve = [&](size_t bytes) {
    char* r = p;
    p += (bytes + 255) & ~(size_t)255;
    return (void*)r;
  };
  half_t* Qh = (half_t*)carve((size_t)BATCH * NPIX * CX * sizeof(half_t));
  half_t* Kh = (half_t*)carve((size_t)BATCH * MPIX * CX * sizeof(half_t));
  half_t* Vt = (half_t*)carve((size_t)BATCH * MPIX * CX * sizeof(half_t));
  half_t* Wq = (half_t*)carve((size_t)CX * CX * sizeof(half_t));
  half_t* Wk = (half_t*)carve((size_t)CX * CY * sizeof(half_t));
  half_t* Wv = (half_t*)carve((size_t)CX * CY * sizeof(half_t));
  float*  Bq = (float*)carve(CX * sizeof(float));
  float*  Bk = (float*)carve(CX * sizeof(float));
  float*  Bv = (float*)carve(CX * sizeof(float));

  fuse_bn_kernel<<<(CX * CX + 255) / 256, 256, 0, stream>>>(w_q, b_q, gq, bq, mq, vq, CX, CX, Wq, Bq);
  fuse_bn_kernel<<<(CX * CY + 255) / 256, 256, 0, stream>>>(w_k, b_k, gk, bk, mk, vk, CX, CY, Wk, Bk);
  fuse_bn_kernel<<<(CX * CY + 255) / 256, 256, 0, stream>>>(w_v, b_v, gv, bv, mv, vv, CX, CY, Wv, Bv);

  proj_kernel<<<BATCH * NPIX / 16, 32, 0, stream>>>(x, Wq, Bq, Qh, CX, NPIX, 0);
  proj_kernel<<<BATCH * MPIX / 16, 32, 0, stream>>>(y, Wk, Bk, Kh, CY, MPIX, 0);
  proj_kernel<<<BATCH * MPIX / 16, 32, 0, stream>>>(y, Wv, Bv, Vt, CY, MPIX, 1);

  attn_kernel<<<BATCH * (NPIX / 128), 256, 0, stream>>>(Qh, Kh, Vt, x, (float*)d_out);
}
